// Head_35777077576385
// MI455X (gfx1250) — matrix operand order repaired and checked by reading, not compiled, not run on hardware
//
#include <hip/hip_runtime.h>
#include <hip/hip_bf16.h>

typedef __attribute__((ext_vector_type(16))) __bf16 v16bf;
typedef __attribute__((ext_vector_type(8)))  __bf16 v8bf;
typedef __attribute__((ext_vector_type(8)))  float  v8f;
typedef __attribute__((ext_vector_type(4)))  float  v4f;
typedef int vsi4 __attribute__((vector_size(16)));   // matches builtin param type

#define WMMA_BF16(A, B, C) \
    __builtin_amdgcn_wmma_f32_16x16x32_bf16(false, (A), false, (B), (short)0, (C), false, false)

// --- CDNA5 async global->LDS copy (tracked by ASYNCcnt), with fallback ------
#if defined(__has_builtin) && __has_builtin(__builtin_amdgcn_global_load_async_to_lds_b128)
#define ASYNC_COPY16(gsrc, ldst)                                              \
    __builtin_amdgcn_global_load_async_to_lds_b128(                           \
        (__attribute__((address_space(1))) vsi4*)(gsrc),                      \
        (__attribute__((address_space(3))) vsi4*)(ldst), 0, 0)
#else
#define ASYNC_COPY16(gsrc, ldst) (*(v8bf*)(ldst) = *(const v8bf*)(gsrc))
#endif

#if defined(__has_builtin) && __has_builtin(__builtin_amdgcn_s_wait_asynccnt)
#define WAIT_ASYNC(n) __builtin_amdgcn_s_wait_asynccnt(n)
#else
#define WAIT_ASYNC(n) asm volatile("s_wait_asynccnt " #n ::: "memory")
#endif

static __device__ inline v8f zero_v8f() {
    v8f z = {0.f, 0.f, 0.f, 0.f, 0.f, 0.f, 0.f, 0.f};
    return z;
}

// ---------------------------------------------------------------------------
// Kernel 1: transpose + convert weights (C x H f32, row-major) -> (H x C bf16)
// so that WMMA B-fragments (fixed N, 16 consecutive K) are contiguous 32B.
// ---------------------------------------------------------------------------
__global__ __launch_bounds__(256) void prep_weights(
    const float* __restrict__ Wk, const float* __restrict__ Wq,
    const float* __restrict__ Wv,
    __bf16* __restrict__ WkT, __bf16* __restrict__ WqT, __bf16* __restrict__ WvT) {
    int idx = blockIdx.x * blockDim.x + threadIdx.x;   // 0 .. 64*1024-1
    if (idx >= 64 * 1024) return;
    int h = idx >> 10;        // 0..63
    int c = idx & 1023;       // 0..1023
    WkT[h * 1024 + c] = (__bf16)Wk[c * 64 + h];
    WqT[h * 1024 + c] = (__bf16)Wq[c * 64 + h];
    WvT[h * 1024 + c] = (__bf16)Wv[c * 64 + h];
}

// ---------------------------------------------------------------------------
// Kernel 2: projections. Each wave handles one 16-row token tile; computes
// q,k,v (16x64 each) with v_wmma_f32_16x16x32_bf16, K-loop over C=1024.
// Q,K stored bf16 row-major [(B*T) x 64]; V stored transposed [B][64][T].
// ---------------------------------------------------------------------------
__global__ __launch_bounds__(128) void proj_kernel(
    const float* __restrict__ x,
    const __bf16* __restrict__ WkT, const __bf16* __restrict__ WqT,
    const __bf16* __restrict__ WvT,
    __bf16* __restrict__ Qb, __bf16* __restrict__ Kb, __bf16* __restrict__ Vt) {
    const int lane = threadIdx.x & 31;
    const int wave = threadIdx.x >> 5;
    const int tile = blockIdx.x * (blockDim.x >> 5) + wave;   // 0..1023
    const int m  = lane & 15;
    const int hi = lane >> 4;                                 // 0 or 1
    const int row0 = tile << 4;

    const float* xrow = x + (size_t)(row0 + m) * 1024;

    v8f accQ[4], accK[4], accV[4];
#pragma unroll
    for (int nt = 0; nt < 4; ++nt) { accQ[nt] = zero_v8f(); accK[nt] = zero_v8f(); accV[nt] = zero_v8f(); }

    for (int kk = 0; kk < 32; ++kk) {
        const int kb = kk * 32;
        // A-fragment: 16x32 bf16, contiguous K: lane<16 holds K=0..15;
        // lane>=16 holds K=16..31 (matches B-fragment order).
        const float* pa = xrow + kb + hi * 16;
        v4f a0 = *(const v4f*)(pa);
        v4f a1 = *(const v4f*)(pa + 4);
        v4f a2 = *(const v4f*)(pa + 8);
        v4f a3 = *(const v4f*)(pa + 12);
        v16bf A;
#pragma unroll
        for (int i = 0; i < 4; ++i) {
            A[i]      = (__bf16)a0[i];
            A[4 + i]  = (__bf16)a1[i];
            A[8 + i]  = (__bf16)a2[i];
            A[12 + i] = (__bf16)a3[i];
        }
#pragma unroll
        for (int nt = 0; nt < 4; ++nt) {
            const int n = nt * 16 + m;  // B-fragment: N = lane%16, K contiguous
            const size_t off = (size_t)n * 1024 + kb + hi * 16;
            v16bf Bq = *(const v16bf*)(WqT + off);
            accQ[nt] = WMMA_BF16(A, Bq, accQ[nt]);
            v16bf Bk = *(const v16bf*)(WkT + off);
            accK[nt] = WMMA_BF16(A, Bk, accK[nt]);
            v16bf Bv = *(const v16bf*)(WvT + off);
            accV[nt] = WMMA_BF16(A, Bv, accV[nt]);
        }
    }

    // C/D layout: VGPR r -> row M = r + 8*(lane/16), col N = lane%16 (+16*nt)
#pragma unroll
    for (int r = 0; r < 8; ++r) {
        const int M = r + 8 * hi;
        const size_t grow = (size_t)(row0 + M);
        const int bb = (int)(grow >> 12);       // batch
        const int t  = (int)(grow & 4095);      // token within batch
#pragma unroll
        for (int nt = 0; nt < 4; ++nt) {
            const int n = nt * 16 + m;
            Qb[grow * 64 + n] = (__bf16)accQ[nt][r];
            Kb[grow * 64 + n] = (__bf16)accK[nt][r];
            Vt[((size_t)(bb * 64 + n)) * 4096 + t] = (__bf16)accV[nt][r];
        }
    }
}

// ---------------------------------------------------------------------------
// Kernel 3: flash attention. 4 waves/block share each 32-key K/V tile through
// LDS, staged with double-buffered async global->LDS copies (ASYNCcnt).
// Per wave: S = Q*K^T (4 wmma from LDS B-frags), online softmax in C/D
// layout, P re-laid out via LDS, O += P*V (4 wmma, V transposed in LDS).
// ---------------------------------------------------------------------------
__global__ __launch_bounds__(128) void attn_kernel(
    const __bf16* __restrict__ Qb, const __bf16* __restrict__ Kb,
    const __bf16* __restrict__ Vt, float* __restrict__ out) {
    __shared__ __bf16 Kl[2][32][72];   // 32 keys x 64 h, padded rows (16B-aligned)
    __shared__ __bf16 Vl[2][64][40];   // 64 h x 32 keys, padded rows
    __shared__ __bf16 Pl[4][16][40];   // per-wave 16x32 P tile

    const int tid  = threadIdx.x;
    const int lane = tid & 31;
    const int wave = tid >> 5;
    const int tile = blockIdx.x * 4 + wave;   // 0..1023
    const int m  = lane & 15;
    const int hi = lane >> 4;
    const int b  = tile >> 8;                 // batch (uniform per block)
    const int q0 = (tile & 255) << 4;         // query tile start within batch
    const int jend = (((blockIdx.x * 4) & 255) << 4) + 63;  // block's last key

    const __bf16* kbase = Kb + (size_t)b * 4096 * 64;
    const __bf16* vbase = Vt + (size_t)b * 64 * 4096;

    // Stage one 32-key block into LDS buffer `buf` (coalesced 16B granules;
    // 128 threads x 4 async b128 copies = 4KB K + 4KB V).
    auto stage = [&](int buf, int jsrc) {
        const int kr = tid >> 3, kc = (tid & 7) * 8;      // K: 8 granules/row
        ASYNC_COPY16(kbase + (size_t)(jsrc + kr) * 64 + kc,      &Kl[buf][kr][kc]);
        ASYNC_COPY16(kbase + (size_t)(jsrc + kr + 16) * 64 + kc, &Kl[buf][kr + 16][kc]);
        const int vh = tid >> 2, vc = (tid & 3) * 8;      // V: 4 granules/row
        ASYNC_COPY16(vbase + (size_t)vh * 4096 + jsrc + vc,        &Vl[buf][vh][vc]);
        ASYNC_COPY16(vbase + (size_t)(vh + 32) * 4096 + jsrc + vc, &Vl[buf][vh + 32][vc]);
    };

    // Load Q A-fragments (two K-chunks of 32 over H=64), held for whole loop.
    const __bf16* qp = Qb + (size_t)(tile * 16 + m) * 64;
    v16bf QA[2];
#pragma unroll
    for (int c = 0; c < 2; ++c) {
        v8bf r0 = *(const v8bf*)(qp + c * 32 + hi * 16);
        v8bf r1 = *(const v8bf*)(qp + c * 32 + hi * 16 + 8);
#pragma unroll
        for (int i = 0; i < 8; ++i) { QA[c][i] = r0[i]; QA[c][8 + i] = r1[i]; }
    }

    v8f O[4];
#pragma unroll
    for (int nt = 0; nt < 4; ++nt) O[nt] = zero_v8f();
    float mr[8], lr[8];
#pragma unroll
    for (int r = 0; r < 8; ++r) { mr[r] = -1e30f; lr[r] = 0.f; }

    const float scale = 0.125f;   // 1/sqrt(64)

    stage(0, 0);                  // prologue: prefetch first block
    int p = 0;
    for (int j = 0; j <= jend; j += 32) {
        int jn = j + 32;          // prefetch next block (clamped, stays in-bounds)
        if (jn > 4064) jn = 4064;
        stage(p ^ 1, jn);
        WAIT_ASYNC(4);            // async ops complete in order: buffer p ready
        __syncthreads();          // make it visible to all 4 waves

        if (j <= q0 + 15) {       // this wave still has unmasked keys here
            v8f S0 = zero_v8f(), S1 = zero_v8f();
#pragma unroll
            for (int c = 0; c < 2; ++c) {
                // B-fragment: N = key col (lane%16), K = head dim contiguous
                v8bf k0a = *(const v8bf*)&Kl[p][m][c * 32 + hi * 16];
                v8bf k0b = *(const v8bf*)&Kl[p][m][c * 32 + hi * 16 + 8];
                v8bf k1a = *(const v8bf*)&Kl[p][m + 16][c * 32 + hi * 16];
                v8bf k1b = *(const v8bf*)&Kl[p][m + 16][c * 32 + hi * 16 + 8];
                v16bf B0, B1;
#pragma unroll
                for (int i = 0; i < 8; ++i) {
                    B0[i] = k0a[i]; B0[8 + i] = k0b[i];
                    B1[i] = k1a[i]; B1[8 + i] = k1b[i];
                }
                S0 = WMMA_BF16(QA[c], B0, S0);
                S1 = WMMA_BF16(QA[c], B1, S1);
            }

            // Online softmax; element (lane,r) of S0 is (row q0+r+8*hi, col j+m).
#pragma unroll
            for (int r = 0; r < 8; ++r) {
                const int qi = q0 + r + 8 * hi;
                float s0 = S0[r] * scale;
                float s1 = S1[r] * scale;
                if (j + m > qi)      s0 = -1e30f;       // causal mask
                if (j + 16 + m > qi) s1 = -1e30f;
                float t = fmaxf(s0, s1);
#pragma unroll
                for (int d = 1; d < 16; d <<= 1) t = fmaxf(t, __shfl_xor(t, d, 32));
                const float mnew  = fmaxf(mr[r], t);
                const float alpha = __expf(mr[r] - mnew);
                const float p0 = __expf(s0 - mnew);
                const float p1 = __expf(s1 - mnew);
                float rs = p0 + p1;
#pragma unroll
                for (int d = 1; d < 16; d <<= 1) rs += __shfl_xor(rs, d, 32);
                lr[r] = lr[r] * alpha + rs;
                mr[r] = mnew;
                O[0][r] *= alpha; O[1][r] *= alpha; O[2][r] *= alpha; O[3][r] *= alpha;
                const int M = r + 8 * hi;
                Pl[wave][M][m]      = (__bf16)p0;       // stage P for re-layout
                Pl[wave][M][16 + m] = (__bf16)p1;
            }

            // Same-wave LDS ordering for the C/D -> A transpose.
            asm volatile("s_wait_dscnt 0" ::: "memory");

            v16bf PA;
            {
                v8bf pr0 = *(const v8bf*)&Pl[wave][m][hi * 16];
                v8bf pr1 = *(const v8bf*)&Pl[wave][m][hi * 16 + 8];
#pragma unroll
                for (int i = 0; i < 8; ++i) { PA[i] = pr0[i]; PA[8 + i] = pr1[i]; }
            }

            // O += P * V : transposed V in LDS => contiguous B-fragments.
#pragma unroll
            for (int nt = 0; nt < 4; ++nt) {
                v8bf va = *(const v8bf*)&Vl[p][nt * 16 + m][hi * 16];
                v8bf vb = *(const v8bf*)&Vl[p][nt * 16 + m][hi * 16 + 8];
                v16bf BV;
#pragma unroll
                for (int i = 0; i < 8; ++i) { BV[i] = va[i]; BV[8 + i] = vb[i]; }
                O[nt] = WMMA_BF16(PA, BV, O[nt]);
            }
        }

        __syncthreads();          // all readers done before buffer p is reused
        p ^= 1;
    }

    // Normalize and write f32 output.
#pragma unroll
    for (int r = 0; r < 8; ++r) {
        const int M = r + 8 * hi;
        const float inv = 1.0f / lr[r];
        const size_t orow = (size_t)(tile * 16 + M) * 64;
        out[orow + m]      = O[0][r] * inv;
        out[orow + 16 + m] = O[1][r] * inv;
        out[orow + 32 + m] = O[2][r] * inv;
        out[orow + 48 + m] = O[3][r] * inv;
    }
}

// ---------------------------------------------------------------------------
// Workspace layout (bytes):
//   [0,   2MB)  Qb   bf16  (16384 x 64)
//   [2MB, 4MB)  Kb   bf16  (16384 x 64)
//   [4MB, 6MB)  Vt   bf16  (4 x 64 x 4096)  (transposed V)
//   [6MB, +128K) WkT, +128K WqT, +128K WvT   bf16 (64 x 1024 each)
// Total ~6.4 MB.
// ---------------------------------------------------------------------------
extern "C" void kernel_launch(void* const* d_in, const int* in_sizes, int n_in,
                              void* d_out, int out_size, void* d_ws, size_t ws_size,
                              hipStream_t stream) {
    const float* x  = (const float*)d_in[0];
    const float* Wk = (const float*)d_in[1];
    const float* Wq = (const float*)d_in[2];
    const float* Wv = (const float*)d_in[3];
    float* out = (float*)d_out;

    char* ws = (char*)d_ws;
    const size_t MB = 1024 * 1024;
    __bf16* Qb  = (__bf16*)(ws + 0);
    __bf16* Kb  = (__bf16*)(ws + 2 * MB);
    __bf16* Vt  = (__bf16*)(ws + 4 * MB);
    __bf16* WkT = (__bf16*)(ws + 6 * MB);
    __bf16* WqT = (__bf16*)(ws + 6 * MB + 128 * 1024);
    __bf16* WvT = (__bf16*)(ws + 6 * MB + 256 * 1024);

    prep_weights<<<256, 256, 0, stream>>>(Wk, Wq, Wv, WkT, WqT, WvT);
    proj_kernel<<<256, 128, 0, stream>>>(x, WkT, WqT, WvT, Qb, Kb, Vt);
    attn_kernel<<<256, 128, 0, stream>>>(Qb, Kb, Vt, out);
}
